// GraphAttentionHead_45681272161009
// MI455X (gfx1250) — matrix operand order repaired and checked by reading, not compiled, not run on hardware
//
#include <hip/hip_runtime.h>

#define IN_DIM 256
#define DK     64
#define NNB    64
#define BATCH  4096

typedef __attribute__((ext_vector_type(16))) _Float16 v16h;
typedef __attribute__((ext_vector_type(8)))  float    v8f;
typedef __attribute__((ext_vector_type(4)))  float    vf4;
typedef __attribute__((ext_vector_type(4)))  int      v4i;

constexpr int LDW  = IN_DIM + 8;  // f16 row stride: 528B rows, 16B aligned, stride-4 bank walk
constexpr int LQ   = DK + 8;      // f16 row stride for 64-wide Q tile
constexpr int LDWF = IN_DIM + 4;  // f32 row stride: 1040B rows, 16B aligned, stride-4 bank walk

#if __has_builtin(__builtin_amdgcn_global_load_async_to_lds_b128) && \
    __has_builtin(__builtin_amdgcn_s_wait_asynccnt)
#define USE_ASYNC_LDS 1
typedef __attribute__((address_space(1))) v4i* glb_v4i;   // global src (prints as __device__)
typedef __attribute__((address_space(3))) v4i* lds_v4i;   // LDS dst    (prints as __shared__)
#endif

__device__ __forceinline__ v8f wmma_f16(v16h a, v16h b, v8f c) {
  return __builtin_amdgcn_wmma_f32_16x16x32_f16(false, a, false, b, (short)0, c, false, false);
}

// A fragment: 16x32 (MxK) f16 tile, row-major, stride lda (halves). ISA 7.12.2 layout.
__device__ __forceinline__ v16h load_a_f16(const _Float16* A, int lda, int lane) {
  const int m = lane & 15, grp = lane >> 4;
  const _Float16* row = A + m * lda;
  v16h f;
#pragma unroll
  for (int v = 0; v < 8; ++v) {
    const int k0 = (v < 4 ? 2 * v : 2 * (v - 4) + 16) + 8 * grp;
    f[2 * v]     = row[k0];
    f[2 * v + 1] = row[k0 + 1];
  }
  return f;
}

// A fragment from an f32 tile (convert on load; 8 consecutive floats -> 2x ds_load_b128).
__device__ __forceinline__ v16h load_a_f32(const float* A, int lda, int lane) {
  const int m = lane & 15, grp = lane >> 4;
  const float* row = A + m * lda;
  v16h f;
#pragma unroll
  for (int v = 0; v < 8; ++v) {
    const int k0 = (v < 4 ? 2 * v : 2 * (v - 4) + 16) + 8 * grp;
    f[2 * v]     = (_Float16)row[k0];
    f[2 * v + 1] = (_Float16)row[k0 + 1];
  }
  return f;
}

// B fragment where logical B[k][n] = W[n][k]; W row-major NxK, stride ldw.
__device__ __forceinline__ v16h load_bT_f16(const _Float16* W, int ldw, int lane) {
  const int n = lane & 15, grp = lane >> 4;
  const _Float16* row = W + n * ldw;
  v16h f;
#pragma unroll
  for (int v = 0; v < 8; ++v) {
    const int k = (v < 4 ? 2 * v : 2 * (v - 4) + 16) + 8 * grp;
    f[2 * v]     = row[k];
    f[2 * v + 1] = row[k + 1];
  }
  return f;
}

// B fragment read + f32->f16 convert from global (small L2-resident Wk).
__device__ __forceinline__ v16h load_b_f32(const float* B, int ldb, int lane) {
  const int n = lane & 15, grp = lane >> 4;
  v16h f;
#pragma unroll
  for (int v = 0; v < 8; ++v) {
    const int k = (v < 4 ? 2 * v : 2 * (v - 4) + 16) + 8 * grp;
    f[2 * v]     = (_Float16)B[k * ldb + n];
    f[2 * v + 1] = (_Float16)B[(k + 1) * ldb + n];
  }
  return f;
}

// ---------------------------------------------------------------------------
// Kernel 1: g[m,:] = (x@Wq^T + bq)[m,:] @ Wk ;  h[m] = q[m,:]·bk ;
// block 0 additionally converts Wv -> f16 once (ws) for the streaming kernel.
// ---------------------------------------------------------------------------
__global__ void __launch_bounds__(128)
k_precompute_g_h(const float* __restrict__ x,
                 const float* __restrict__ Wq, const float* __restrict__ bq,
                 const float* __restrict__ Wk, const float* __restrict__ bk,
                 const float* __restrict__ Wv, _Float16* __restrict__ Wv16,
                 float* __restrict__ g, float* __restrict__ h) {
  __shared__ __align__(16) _Float16 sWq[DK * LDW];
  __shared__ __align__(16) _Float16 sX[16 * LDW];
  __shared__ __align__(16) _Float16 sQ[16 * LQ];

  const int tid  = threadIdx.x;
  const int lane = tid & 31;
  const int wave = tid >> 5;
  const int row0 = blockIdx.x * 16;

  if (blockIdx.x == 0)
    for (int idx = tid; idx < DK * IN_DIM; idx += 128)
      Wv16[idx] = (_Float16)Wv[idx];

  for (int idx = tid; idx < 16 * IN_DIM; idx += 128) {
    const int r = idx >> 8, d = idx & (IN_DIM - 1);
    sX[r * LDW + d] = (_Float16)x[(size_t)(row0 + r) * IN_DIM + d];
  }
  for (int idx = tid; idx < DK * IN_DIM; idx += 128) {
    const int r = idx >> 8, d = idx & (IN_DIM - 1);
    sWq[r * LDW + d] = (_Float16)Wq[(size_t)r * IN_DIM + d];
  }
  __syncthreads();

  {  // Q[16,64] = xTile @ Wq^T + bq
    const int n0 = wave * 16;
    v8f acc = {};
#pragma unroll
    for (int kc = 0; kc < IN_DIM / 32; ++kc) {
      const v16h a = load_a_f16(sX + kc * 32, LDW, lane);
      const v16h b = load_bT_f16(sWq + n0 * LDW + kc * 32, LDW, lane);
      acc = wmma_f16(a, b, acc);
    }
    const int n = n0 + (lane & 15), grp = lane >> 4;
    const float bqv = bq[n];
#pragma unroll
    for (int v = 0; v < 8; ++v)
      sQ[(v + 8 * grp) * LQ + n] = (_Float16)(acc[v] + bqv);
  }
  __syncthreads();

  for (int nt = wave; nt < IN_DIM / 16; nt += 4) {  // G[16,256] = Q @ Wk
    v8f acc = {};
#pragma unroll
    for (int kc = 0; kc < DK / 32; ++kc) {
      const v16h a = load_a_f16(sQ + kc * 32, LQ, lane);
      const v16h b = load_b_f32(Wk + (size_t)(kc * 32) * IN_DIM + nt * 16, IN_DIM, lane);
      acc = wmma_f16(a, b, acc);
    }
    const int n = nt * 16 + (lane & 15), grp = lane >> 4;
#pragma unroll
    for (int v = 0; v < 8; ++v)
      g[(size_t)(row0 + v + 8 * grp) * IN_DIM + n] = acc[v];
  }

  if (tid < 16) {
    float acc = 0.f;
#pragma unroll
    for (int k = 0; k < DK; ++k) acc += (float)sQ[tid * LQ + k] * bk[k];
    h[row0 + tid] = acc;
  }
}

// ===========================================================================
// Variant B: single pass over xq. 64-row tiles staged f32 into LDS (async
// global->LDS when available); every wave does 2 WMMA V-tiles (B-fragments
// register-cached) + 8 f32 score dots. Contiguous 8KB f16 V write-out.
// ===========================================================================
__global__ void __launch_bounds__(256)
k_scores_v(const float* __restrict__ xq, const float* __restrict__ g,
           const float* __restrict__ h, const _Float16* __restrict__ Wv16,
           float* __restrict__ S, _Float16* __restrict__ V) {
  __shared__ __align__(16) float    sXq[64 * LDWF];   // 66.6 KB, f32 tile
  __shared__ __align__(16) _Float16 sWv[DK * LDW];    // 33.8 KB
  __shared__ __align__(16) _Float16 sV[64 * DK];      // 8 KB copy-out tile

  const int tid  = threadIdx.x;
  const int lane = tid & 31;
  const int wave = tid >> 5;
  const int t0 = blockIdx.x * 64;          // 64 rows, one slab (4096 % 64 == 0)
  const int a  = t0 >> 12;                 // slab r
  const int m0 = t0 & (BATCH - 1);

#ifdef USE_ASYNC_LDS
  // gfx1250 async DMA: global -> LDS, tracked by ASYNCcnt (no VGPR round-trip)
  for (int idx4 = tid * 4; idx4 < 64 * IN_DIM; idx4 += 256 * 4) {
    const int r = idx4 >> 8, d = idx4 & (IN_DIM - 1);
    __builtin_amdgcn_global_load_async_to_lds_b128(
        (glb_v4i)(xq + (size_t)(t0 + r) * IN_DIM + d),
        (lds_v4i)(sXq + r * LDWF + d), 0, 0);
  }
#else
  for (int idx = tid; idx < 64 * IN_DIM; idx += 256) {
    const int r = idx >> 8, d = idx & (IN_DIM - 1);
    sXq[r * LDWF + d] =
        __builtin_nontemporal_load(xq + (size_t)(t0 + r) * IN_DIM + d);
  }
#endif
  {  // stage pre-converted Wv f16 as dwords
    const unsigned* src = (const unsigned*)Wv16;
    unsigned* dst = (unsigned*)sWv;
    for (int idx = tid; idx < DK * (IN_DIM / 2); idx += 256) {
      const int r = idx >> 7, du = idx & (IN_DIM / 2 - 1);
      dst[r * (LDW / 2) + du] = src[r * (IN_DIM / 2) + du];
    }
  }
#ifdef USE_ASYNC_LDS
  __builtin_amdgcn_s_wait_asynccnt(0);
#endif
  __syncthreads();

  {  // V tiles: wave -> (nt = wave&3, row-tiles rt0*2, rt0*2+1); B-frags in VGPRs
    const int nt = wave & 3;
    const int rt0 = wave >> 2;
    v16h bf[8];
#pragma unroll
    for (int kc = 0; kc < IN_DIM / 32; ++kc)
      bf[kc] = load_bT_f16(sWv + (nt * 16) * LDW + kc * 32, LDW, lane);
#pragma unroll
    for (int rr = 0; rr < 2; ++rr) {
      const int rt = rt0 * 2 + rr;
      v8f acc = {};
#pragma unroll
      for (int kc = 0; kc < IN_DIM / 32; ++kc)
        acc = wmma_f16(load_a_f32(sXq + (rt * 16) * LDWF + kc * 32, LDWF, lane),
                       bf[kc], acc);
      const int n = nt * 16 + (lane & 15), grp = lane >> 4;
#pragma unroll
      for (int v = 0; v < 8; ++v)
        sV[(rt * 16 + v + 8 * grp) * DK + n] = (_Float16)acc[v];
    }
  }

  // score dots (full f32): each wave 8 rows
  for (int rl = 0; rl < 8; ++rl) {
    const int r = wave * 8 + rl;
    const int m = m0 + r;
    const float* xr = sXq + r * LDWF + lane * 8;
    const float* gr = g + (size_t)m * IN_DIM + lane * 8;
    float s = 0.f;
#pragma unroll
    for (int d = 0; d < 8; ++d) s += xr[d] * gr[d];
#pragma unroll
    for (int off = 16; off > 0; off >>= 1) s += __shfl_xor(s, off, 32);
    if (lane == 0) {
      const int i = m >> 6, c = m & 63;
      S[(size_t)i * BATCH + (c << 6) + a] = (s + h[m]) * 0.125f;
    }
  }
  __syncthreads();

  // contiguous 8KB store: rows t0..t0+63 of V are adjacent in memory
#pragma unroll
  for (int p = 0; p < 4; ++p)
    ((uint2*)(V + (size_t)t0 * DK))[p * 256 + tid] = ((const uint2*)sV)[p * 256 + tid];
}

__global__ void __launch_bounds__(256)
k_out_v(const _Float16* __restrict__ V, const float* __restrict__ S,
        const float* __restrict__ bv, float* __restrict__ out) {
  __shared__ float sw[16][NNB];
  const int tid = threadIdx.x;
  const int b0 = blockIdx.x * 16;

  if (tid < 16) {
    const int b = b0 + tid;
    float mx = -3.4e38f;
#pragma unroll
    for (int i = 0; i < NNB; ++i) {
      const float v = S[(size_t)i * BATCH + b];
      sw[tid][i] = v;
      mx = fmaxf(mx, v);
    }
    float sum = 0.f;
#pragma unroll
    for (int i = 0; i < NNB; ++i) {
      const float e = __expf(sw[tid][i] - mx);
      sw[tid][i] = e;
      sum += e;
    }
    const float inv = 1.f / sum;
#pragma unroll
    for (int i = 0; i < NNB; ++i) sw[tid][i] *= inv;
  }
  __syncthreads();

  const int d  = tid & (DK - 1);
  const int bq = tid >> 6;
  const float bvd = bv[d];
#pragma unroll
  for (int p = 0; p < 4; ++p) {
    const int bl = p * 4 + bq;
    const int b  = b0 + bl;
    float acc = 0.f;
#pragma unroll 8
    for (int i = 0; i < NNB; ++i)
      acc += sw[bl][i] * (float)V[((size_t)i * BATCH + b) * DK + d];
    out[(size_t)b * DK + d] = acc + bvd;
  }
}

// ===========================================================================
// Variant A (fallback for small ws): two NT streaming passes over xq.
// ===========================================================================
__global__ void __launch_bounds__(256)
k_scores(const float* __restrict__ xq, const float* __restrict__ g,
         const float* __restrict__ h, float* __restrict__ S) {
  const int lane = threadIdx.x & 31;
  const int wave = threadIdx.x >> 5;
  const int t = blockIdx.x * 8 + wave;
  const int a = t >> 12;
  const int m = t & (BATCH - 1);

  const vf4 x0 = __builtin_nontemporal_load((const vf4*)(xq + (size_t)t * IN_DIM + lane * 8));
  const vf4 x1 = __builtin_nontemporal_load((const vf4*)(xq + (size_t)t * IN_DIM + lane * 8 + 4));
  const vf4 g0 = *(const vf4*)(g + (size_t)m * IN_DIM + lane * 8);
  const vf4 g1 = *(const vf4*)(g + (size_t)m * IN_DIM + lane * 8 + 4);
  float s = x0[0]*g0[0] + x0[1]*g0[1] + x0[2]*g0[2] + x0[3]*g0[3]
          + x1[0]*g1[0] + x1[1]*g1[1] + x1[2]*g1[2] + x1[3]*g1[3];
#pragma unroll
  for (int off = 16; off > 0; off >>= 1) s += __shfl_xor(s, off, 32);
  if (lane == 0) {
    const int i = m >> 6, c = m & 63;
    S[(size_t)i * BATCH + (c << 6) + a] = (s + h[m]) * 0.125f;
  }
}

__global__ void __launch_bounds__(256)
k_softmax_out(const float* __restrict__ xq, const float* __restrict__ S,
              const float* __restrict__ Wv, const float* __restrict__ bv,
              float* __restrict__ out) {
  __shared__ float sw[16][NNB];
  __shared__ __align__(16) _Float16 sXb[16 * LDW];
  __shared__ __align__(16) _Float16 sWv[DK * LDW];

  const int tid  = threadIdx.x;
  const int lane = tid & 31;
  const int wave = tid >> 5;
  const int b0 = blockIdx.x * 16;

  for (int idx = tid; idx < DK * IN_DIM; idx += 256) {
    const int r = idx >> 8, d = idx & (IN_DIM - 1);
    sWv[r * LDW + d] = (_Float16)Wv[(size_t)r * IN_DIM + d];
  }

  if (tid < 16) {
    const int b = b0 + tid;
    float mx = -3.4e38f;
#pragma unroll
    for (int i = 0; i < NNB; ++i) {
      const float v = S[(size_t)i * BATCH + b];
      sw[tid][i] = v;
      mx = fmaxf(mx, v);
    }
    float sum = 0.f;
#pragma unroll
    for (int i = 0; i < NNB; ++i) {
      const float e = __expf(sw[tid][i] - mx);
      sw[tid][i] = e;
      sum += e;
    }
    const float inv = 1.f / sum;
#pragma unroll
    for (int i = 0; i < NNB; ++i) sw[tid][i] *= inv;
  }
  __syncthreads();

  for (int bl = 0; bl < 16; ++bl) {
    const size_t base = (size_t)(b0 + bl) * IN_DIM + tid;
    float acc = 0.f;
#pragma unroll 8
    for (int i = 0; i < NNB; ++i)
      acc += sw[bl][i] * __builtin_nontemporal_load(xq + base + (size_t)i * BATCH * IN_DIM);
    sXb[bl * LDW + tid] = (_Float16)acc;
  }
  __syncthreads();

  if (wave < 4) {
    const int n0 = wave * 16;
    v8f acc = {};
#pragma unroll
    for (int kc = 0; kc < IN_DIM / 32; ++kc) {
      const v16h af = load_a_f16(sXb + kc * 32, LDW, lane);
      const v16h bm = load_bT_f16(sWv + n0 * LDW + kc * 32, LDW, lane);
      acc = wmma_f16(af, bm, acc);
    }
    const int n = n0 + (lane & 15), grp = lane >> 4;
    const float bvv = bv[n];
#pragma unroll
    for (int v = 0; v < 8; ++v)
      out[(size_t)(b0 + v + 8 * grp) * DK + n] = acc[v] + bvv;
  }
}

extern "C" void kernel_launch(void* const* d_in, const int* in_sizes, int n_in,
                              void* d_out, int out_size, void* d_ws, size_t ws_size,
                              hipStream_t stream) {
  (void)in_sizes; (void)n_in; (void)out_size;
  const float* x  = (const float*)d_in[0];
  const float* xq = (const float*)d_in[1];
  const float* Wq = (const float*)d_in[2];
  const float* bq = (const float*)d_in[3];
  const float* Wk = (const float*)d_in[4];
  const float* bk = (const float*)d_in[5];
  const float* Wv = (const float*)d_in[6];
  const float* bv = (const float*)d_in[7];
  float* out = (float*)d_out;

  // ws: g[4096*256] | h[4096] | S[64*4096] (f32) | V[64*4096*64] | Wv16[64*256] (f16)
  float* g = (float*)d_ws;
  float* h = g + (size_t)BATCH * IN_DIM;
  float* S = h + BATCH;
  _Float16* V    = (_Float16*)(S + (size_t)NNB * BATCH);
  _Float16* Wv16 = V + (size_t)NNB * BATCH * DK;

  const size_t need_f32 = ((size_t)BATCH * IN_DIM + BATCH + (size_t)NNB * BATCH) * sizeof(float);
  const size_t need_B   = need_f32 +
      ((size_t)NNB * BATCH * DK + (size_t)DK * IN_DIM) * sizeof(_Float16);

  if (ws_size >= need_B) {
    k_precompute_g_h<<<BATCH / 16, 128, 0, stream>>>(x, Wq, bq, Wk, bk, Wv, Wv16, g, h);
    k_scores_v<<<(NNB * BATCH) / 64, 256, 0, stream>>>(xq, g, h, Wv16, S, V);
    k_out_v<<<BATCH / 16, 256, 0, stream>>>(V, S, bv, out);
  } else {
    k_precompute_g_h<<<BATCH / 16, 128, 0, stream>>>(x, Wq, bq, Wk, bk, Wv, Wv16, g, h);
    k_scores<<<(NNB * BATCH) / 8, 256, 0, stream>>>(xq, g, h, S);
    k_softmax_out<<<BATCH / 16, 256, 0, stream>>>(xq, S, Wv, bv, out);
  }
}
